// FCOSPrototype_8967891714138
// MI455X (gfx1250) — matrix operand order repaired and checked by reading, not compiled, not run on hardware
//
#include <hip/hip_runtime.h>
#include <hip/hip_bf16.h>

typedef __attribute__((ext_vector_type(16))) _Float16 v16h;
typedef __attribute__((ext_vector_type(8)))  _Float16 v8h;
typedef __attribute__((ext_vector_type(8)))  float    v8f;

#define CATS      17
#define SCALES    5
#define NSEG      85            // CATS * SCALES
#define DIM       256
#define HALF_DIM  128
#define BX        128           // row-partition blocks
#define SLICE_FLTS (NSEG * HALF_DIM)   // 10880 floats per partial slice

// ---------------------------------------------------------------------------
// K1: streaming segment-sum. grid = (BX, 2). blockIdx.y selects dim half.
// Each block accumulates its row subset into LDS (ds_add_f32), then writes one
// partial slice (85 x 128 f32) to scratch. y==0 blocks also count rows/segment.
// ---------------------------------------------------------------------------
__global__ void seg_partial_kernel(const float* __restrict__ feats,
                                   const int*   __restrict__ tgt,
                                   const int*   __restrict__ lvl,
                                   float* __restrict__ psums,
                                   float* __restrict__ pcnts,
                                   int n) {
    __shared__ float acc[SLICE_FLTS];
    __shared__ float cnt[NSEG];
    const int tid = threadIdx.x;
    for (int i = tid; i < SLICE_FLTS; i += 256) acc[i] = 0.f;
    if (tid < NSEG) cnt[tid] = 0.f;
    __syncthreads();

    const int y   = blockIdx.y;     // dim half
    const int d   = tid & 127;      // dim within half
    const int sub = tid >> 7;       // row within pair (2 rows / iter)

    for (int p = blockIdx.x; ; p += BX) {
        long long r = (long long)p * 2 + sub;
        if (r >= n) break;
        const int seg = tgt[r] * SCALES + lvl[r];
        const float v = feats[r * DIM + y * HALF_DIM + d];
        atomicAdd(&acc[seg * HALF_DIM + d], v);            // ds_add_f32
        if (y == 0 && d == 0) atomicAdd(&cnt[seg], 1.0f);
    }
    __syncthreads();

    float* out = psums + (long long)(blockIdx.x * 2 + y) * SLICE_FLTS;
    for (int i = tid; i < SLICE_FLTS; i += 256) out[i] = acc[i];
    if (y == 0 && tid < NSEG) pcnts[blockIdx.x * NSEG + tid] = cnt[tid];
}

// ---------------------------------------------------------------------------
// K2: reduce partials -> delta, normalize delta & prototypes, stage zero-padded
// 32x256 f16 matrices per scale for WMMA, write mask. grid = 5*32, block = 256.
// ---------------------------------------------------------------------------
__global__ void finalize_kernel(const float* __restrict__ psums,
                                const float* __restrict__ pcnts,
                                const float* __restrict__ protos,
                                _Float16* __restrict__ v1h,
                                _Float16* __restrict__ v2h,
                                float* __restrict__ maskv) {
    const int c   = blockIdx.x & 31;
    const int s   = blockIdx.x >> 5;
    const int tid = threadIdx.x;               // == d in [0,256)
    _Float16* prow = v1h + (s * 32 + c) * DIM;
    _Float16* drow = v2h + (s * 32 + c) * DIM;
    if (c >= CATS) {                            // zero padding rows 17..31
        prow[tid] = (_Float16)0.f;
        drow[tid] = (_Float16)0.f;
        return;
    }
    const int row = c * SCALES + s;             // == segment id

    __shared__ float red[256];
    __shared__ float scnt;

    const int y = tid >> 7, dl = tid & 127;
    float sum = 0.f;
    for (int b = 0; b < BX; ++b)
        sum += psums[(long long)(b * 2 + y) * SLICE_FLTS + row * HALF_DIM + dl];
    if (tid == 0) {
        float cs = 0.f;
        for (int b = 0; b < BX; ++b) cs += pcnts[b * NSEG + row];
        scnt = cs;
    }
    __syncthreads();
    const float cn   = scnt;
    const float delta = (cn > 0.f) ? (sum / cn) : 0.01f;

    // ||delta||
    red[tid] = delta * delta; __syncthreads();
    for (int o = 128; o > 0; o >>= 1) { if (tid < o) red[tid] += red[tid + o]; __syncthreads(); }
    const float nd = sqrtf(red[0]); __syncthreads();
    drow[tid] = (_Float16)(delta / nd);

    // ||prototype row||
    const float p = protos[row * DIM + tid];
    red[tid] = p * p; __syncthreads();
    for (int o = 128; o > 0; o >>= 1) { if (tid < o) red[tid] += red[tid + o]; __syncthreads(); }
    const float np = sqrtf(red[0]);
    prow[tid] = (_Float16)(p / np);

    if (tid == 0) maskv[row] = (cn > 0.f) ? 1.f : 0.f;
}

// ---------------------------------------------------------------------------
// K3: per-scale 17x17 gram matrix via V_WMMA_F32_16X16X32_F16.
// grid = 5 (one wave per scale), block = 32 (EXEC all-ones for WMMA).
// L[m][n] = sum_k P[m,k]*Q[n,k]; 2x2 tiles of 16 cover 32x32 (rows>=17 are 0).
// ---------------------------------------------------------------------------
__global__ void gram_wmma_kernel(const _Float16* __restrict__ v1h,
                                 const _Float16* __restrict__ v2h,
                                 float* __restrict__ logits) {
    const int s    = blockIdx.x;
    const int lane = threadIdx.x;
    const _Float16* P = v1h + s * 32 * DIM;
    const _Float16* Q = v2h + s * 32 * DIM;

    const int ml = lane & 15;
    const int kb = (lane >> 4) * 8;     // A: K sub-base (K = kb..kb+7, 16+kb..16+kb+7)


    v8f acc00 = {}, acc01 = {}, acc10 = {}, acc11 = {};
#pragma unroll
    for (int kc = 0; kc < 8; ++kc) {
        const int ko = kc * 32;
        v8h a0lo = *(const v8h*)(P + (ml     ) * DIM + ko + kb);
        v8h a0hi = *(const v8h*)(P + (ml     ) * DIM + ko + 16 + kb);
        v8h a1lo = *(const v8h*)(P + (16 + ml) * DIM + ko + kb);
        v8h a1hi = *(const v8h*)(P + (16 + ml) * DIM + ko + 16 + kb);
        v16h a0, a1;
#pragma unroll
        for (int i = 0; i < 8; ++i) {
            a0[i] = a0lo[i]; a0[i + 8] = a0hi[i];
            a1[i] = a1lo[i]; a1[i + 8] = a1hi[i];
        }
        v8h b0lo = *(const v8h*)(Q + (ml     ) * DIM + ko + kb);
        v8h b0hi = *(const v8h*)(Q + (ml     ) * DIM + ko + 16 + kb);
        v8h b1lo = *(const v8h*)(Q + (16 + ml) * DIM + ko + kb);
        v8h b1hi = *(const v8h*)(Q + (16 + ml) * DIM + ko + 16 + kb);
        v16h b0, b1;
#pragma unroll
        for (int i = 0; i < 8; ++i) {
            b0[i] = b0lo[i]; b0[i + 8] = b0hi[i];
            b1[i] = b1lo[i]; b1[i + 8] = b1hi[i];
        }
        acc00 = __builtin_amdgcn_wmma_f32_16x16x32_f16(false, a0, false, b0, (short)0, acc00, false, false);
        acc01 = __builtin_amdgcn_wmma_f32_16x16x32_f16(false, a0, false, b1, (short)0, acc01, false, false);
        acc10 = __builtin_amdgcn_wmma_f32_16x16x32_f16(false, a1, false, b0, (short)0, acc10, false, false);
        acc11 = __builtin_amdgcn_wmma_f32_16x16x32_f16(false, a1, false, b1, (short)0, acc11, false, false);
    }

    const float invT = 1.f / 0.07f;
    const int nn = lane & 15;
    const int mb = (lane >> 4) * 8;     // D tile: M = mb + r, N = nn
#pragma unroll
    for (int r = 0; r < 8; ++r) {
        const int m = mb + r;
        if (m < CATS) {
            if (nn < CATS)      logits[(m * SCALES + s) * CATS + nn]      = acc00[r] * invT;
            if (16 + nn < CATS) logits[(m * SCALES + s) * CATS + 16 + nn] = acc01[r] * invT;
        }
        if (16 + m < CATS) {
            if (nn < CATS)      logits[((16 + m) * SCALES + s) * CATS + nn]      = acc10[r] * invT;
            if (16 + nn < CATS) logits[((16 + m) * SCALES + s) * CATS + 16 + nn] = acc11[r] * invT;
        }
    }
}

// ---------------------------------------------------------------------------
// K4: masked InfoNCE mean over 85 rows x 17 cols. 1 block.
// labels[i] = i % 17 (reference's tile(arange(C), S)).
// ---------------------------------------------------------------------------
__global__ void loss_kernel(const float* __restrict__ logits,
                            const float* __restrict__ maskv,
                            float* __restrict__ out) {
    __shared__ float r1[128];
    __shared__ float r2[128];
    const int t = threadIdx.x;
    float val = 0.f, msk = 0.f;
    if (t < NSEG) {
        float x[CATS];
        float mx = -1e30f;
#pragma unroll
        for (int k = 0; k < CATS; ++k) { x[k] = logits[t * CATS + k]; mx = fmaxf(mx, x[k]); }
        float se = 0.f;
#pragma unroll
        for (int k = 0; k < CATS; ++k) se += expf(x[k] - mx);
        const int lbl = t % CATS;
        const float per = mx + logf(se) - x[lbl];  // -log_softmax at label
        msk = maskv[t];
        val = per * msk;
    }
    r1[t] = val; r2[t] = msk; __syncthreads();
    for (int o = 64; o > 0; o >>= 1) {
        if (t < o) { r1[t] += r1[t + o]; r2[t] += r2[t + o]; }
        __syncthreads();
    }
    if (t == 0) out[0] = r1[0] / fmaxf(r2[0], 1.0f);
}

// ---------------------------------------------------------------------------
extern "C" void kernel_launch(void* const* d_in, const int* in_sizes, int n_in,
                              void* d_out, int out_size, void* d_ws, size_t ws_size,
                              hipStream_t stream) {
    const float* feats  = (const float*)d_in[0];
    const int*   tgt    = (const int*)  d_in[1];
    const int*   lvl    = (const int*)  d_in[2];
    const float* protos = (const float*)d_in[3];
    const int n = in_sizes[1];                     // N rows

    float* ws     = (float*)d_ws;
    float* psums  = ws;                                         // BX*2*10880 f32
    float* pcnts  = psums + (size_t)BX * 2 * SLICE_FLTS;        // BX*85 f32
    float* maskv  = pcnts + (size_t)BX * NSEG;                  // 85 f32
    float* logits = maskv + NSEG;                               // 85*17 f32
    size_t fl = (size_t)BX * 2 * SLICE_FLTS + (size_t)BX * NSEG + NSEG + (size_t)NSEG * CATS;
    fl = (fl + 15) & ~(size_t)15;                               // 64B align for f16 arrays
    _Float16* v1h = (_Float16*)(ws + fl);                       // 5*32*256 f16
    _Float16* v2h = v1h + (size_t)SCALES * 32 * DIM;            // 5*32*256 f16

    seg_partial_kernel<<<dim3(BX, 2), 256, 0, stream>>>(feats, tgt, lvl, psums, pcnts, n);
    finalize_kernel<<<SCALES * 32, DIM, 0, stream>>>(psums, pcnts, protos, v1h, v2h, maskv);
    gram_wmma_kernel<<<SCALES, 32, 0, stream>>>(v1h, v2h, logits);
    loss_kernel<<<1, 128, 0, stream>>>(logits, maskv, (float*)d_out);

    (void)n_in; (void)out_size; (void)ws_size;
}
